// Model_52578989637798
// MI455X (gfx1250) — matrix operand order repaired and checked by reading, not compiled, not run on hardware
//
#include <hip/hip_runtime.h>
#include <hip/hip_bf16.h>
#include <math.h>

typedef __bf16 bf16_t;
typedef __attribute__((ext_vector_type(8)))  __bf16 v8bf;
typedef __attribute__((ext_vector_type(16))) __bf16 v16bf;
typedef __attribute__((ext_vector_type(8)))  float  v8f;

#define B_DIM 32
#define L_DIM 512
#define C_DIM 862
#define P_DIM 720
#define P_PAD 768                 // next multiple of 64
#define M_ROWS (B_DIM * C_DIM)    // 27584, divisible by 32

// ---------------------------------------------------------------------------
// Fragment loaders per CDNA5 WMMA VGPR layouts (CDNA5 ISA §7.12.2)
// A 16x32 bf16: lanes 0-15 rows M=0..15 hold K {0..7, 16..23}; lanes 16-31
// hold the same rows with K {8..15, 24..31}.
// B 32x16 bf16 from weights stored N x K row-major: lane group 0 holds
// K {0..7, 16..23} of column n=lane, group 1 holds K {8..15, 24..31}.
// ---------------------------------------------------------------------------
__device__ inline v16bf load_fragA(const bf16_t* __restrict__ p) {
  v8bf lo = *(const v8bf*)(p);
  v8bf hi = *(const v8bf*)(p + 16);
  return __builtin_shufflevector(lo, hi, 0,1,2,3,4,5,6,7,8,9,10,11,12,13,14,15);
}
__device__ inline v16bf load_fragB(const bf16_t* __restrict__ p) {
  v8bf lo = *(const v8bf*)(p);
  v8bf hi = *(const v8bf*)(p + 16);
  return __builtin_shufflevector(lo, hi, 0,1,2,3,4,5,6,7,8,9,10,11,12,13,14,15);
}

__device__ inline void wmma_step(const v16bf (&a)[2], const v16bf (&b)[4],
                                 v8f (&acc)[2][4]) {
#pragma unroll
  for (int nt = 0; nt < 4; ++nt) {
#pragma unroll
    for (int mt = 0; mt < 2; ++mt)
      acc[mt][nt] = __builtin_amdgcn_wmma_f32_16x16x32_bf16(
          false, a[mt], false, b[nt], (short)0, acc[mt][nt], false, false);
  }
}

// ---------------------------------------------------------------------------
// Tiled bf16 GEMM:  C[M x Nreal] = A[M x K] * Bw[Npad x K]^T
// Block = 8 waves (256 threads). Wave tile = 32 x 64 (2 M-frags x 4 N-frags).
// Npad is a multiple of 64 (B zero-padded), so the k-loop is branch-free;
// only the epilogue store is guarded against Nreal. k-loop is double-buffered
// (ping-pong fragments, unroll-by-2; K % 64 == 0 for all stages).
// Epilogues: 0: f32   1: relu->bf16   2: sigmoid->f32
//            3: +bias, transposed (B,P,C) f32 store
// ---------------------------------------------------------------------------
template <int EPI>
__global__ __launch_bounds__(256)
void gemm_bf16_k(const bf16_t* __restrict__ A, const bf16_t* __restrict__ Bw,
                 float* __restrict__ Cf, bf16_t* __restrict__ Cb,
                 int Msz, int Nreal, int K,
                 const float* __restrict__ bias, int Cdim, int Pdim) {
  const int lane = threadIdx.x & 31;
  const int wave = threadIdx.x >> 5;
  const int row0 = blockIdx.x * 256 + wave * 32;
  if (row0 >= Msz) return;                 // wave-uniform (Msz % 32 == 0)
  const int col0 = blockIdx.y * 64;

  const int r   = lane & 15;
  const int sel = lane >> 4;

  // 32-bit per-lane element offsets (A, B are < 2^27 elements).
  const int aOff0 = (row0 + r) * K + sel * 8;
  const int aOff1 = aOff0 + 16 * K;
  int bOff[4];
#pragma unroll
  for (int nt = 0; nt < 4; ++nt) bOff[nt] = (col0 + nt * 16 + r) * K + sel * 8;

  v8f acc[2][4];
#pragma unroll
  for (int mt = 0; mt < 2; ++mt)
#pragma unroll
    for (int nt = 0; nt < 4; ++nt)
#pragma unroll
      for (int e = 0; e < 8; ++e) acc[mt][nt][e] = 0.0f;

  // Ping-pong fragment buffers.
  v16bf aP[2], bP[4], aQ[2], bQ[4];
  aP[0] = load_fragA(A + aOff0);
  aP[1] = load_fragA(A + aOff1);
#pragma unroll
  for (int nt = 0; nt < 4; ++nt) bP[nt] = load_fragB(Bw + bOff[nt]);

  for (int kb = 0; kb < K; kb += 64) {
    // Issue loads for kb+32 while WMMAs consume kb.
    aQ[0] = load_fragA(A + aOff0 + kb + 32);
    aQ[1] = load_fragA(A + aOff1 + kb + 32);
#pragma unroll
    for (int nt = 0; nt < 4; ++nt) bQ[nt] = load_fragB(Bw + bOff[nt] + kb + 32);
    wmma_step(aP, bP, acc);
    if (kb + 64 < K) {
      aP[0] = load_fragA(A + aOff0 + kb + 64);
      aP[1] = load_fragA(A + aOff1 + kb + 64);
#pragma unroll
      for (int nt = 0; nt < 4; ++nt)
        bP[nt] = load_fragB(Bw + bOff[nt] + kb + 64);
    }
    wmma_step(aQ, bQ, acc);
  }

  // C/D layout: VGPR e, lanes 0-15 -> M=e, N=lane; lanes 16-31 -> M=8+e.
#pragma unroll
  for (int mt = 0; mt < 2; ++mt) {
#pragma unroll
    for (int nt = 0; nt < 4; ++nt) {
      const int cb = col0 + nt * 16;
      if (cb >= Nreal) continue;           // uniform (Nreal % 16 == 0)
      const int n = cb + r;
#pragma unroll
      for (int e = 0; e < 8; ++e) {
        const int m = row0 + mt * 16 + sel * 8 + e;
        const float v = acc[mt][nt][e];
        if (EPI == 0) {
          Cf[(size_t)m * Nreal + n] = v;
        } else if (EPI == 1) {
          Cb[(size_t)m * Nreal + n] = (bf16_t)fmaxf(v, 0.0f);
        } else if (EPI == 2) {
          Cf[(size_t)m * Nreal + n] = 1.0f / (1.0f + expf(-v));
        } else {
          const int b = m / Cdim;
          const int c = m - b * Cdim;
          Cf[(size_t)b * Pdim * Cdim + (size_t)n * Cdim + c] = v + bias[n];
        }
      }
    }
  }
}

// ---------------------------------------------------------------------------
// DCT matrix, stored N x K (k-major rows) as the GEMM's B operand, bf16.
// ---------------------------------------------------------------------------
__global__ void gen_dct_k(bf16_t* __restrict__ Dw) {
  const int idx = blockIdx.x * blockDim.x + threadIdx.x;
  if (idx >= L_DIM * L_DIM) return;
  const int k = idx >> 9;
  const int l = idx & (L_DIM - 1);
  const double ang = 3.14159265358979323846 * (2.0 * l + 1.0) * (double)k /
                     (2.0 * L_DIM);
  Dw[idx] = (bf16_t)(2.0f * (float)cos(ang));
}

__global__ void cvt_bf16_k(const float* __restrict__ in, bf16_t* __restrict__ out,
                           int n) {
  const int i = blockIdx.x * blockDim.x + threadIdx.x;
  if (i < n) out[i] = (bf16_t)in[i];
}

// Convert Nreal x K f32 -> Npad x K bf16 with zero row padding.
__global__ void cvt_pad_bf16_k(const float* __restrict__ in,
                               bf16_t* __restrict__ out, int Nreal, int Npad,
                               int K) {
  const int i = blockIdx.x * blockDim.x + threadIdx.x;
  if (i >= Npad * K) return;
  const int nrow = i / K;
  out[i] = (nrow < Nreal) ? (bf16_t)in[i] : (bf16_t)0.0f;
}

// x (B,L,C) f32  ->  A0 (B*C, L) bf16, via 32x32 LDS tile transpose per batch.
__global__ __launch_bounds__(256)
void transpose_x_k(const float* __restrict__ x, bf16_t* __restrict__ A0) {
  __shared__ float tile[32][33];
  const int b  = blockIdx.z;
  const int l0 = blockIdx.x * 32;
  const int c0 = blockIdx.y * 32;
  const int tx = threadIdx.x, ty = threadIdx.y;   // 32 x 8
#pragma unroll
  for (int j = 0; j < 32; j += 8) {
    const int l = l0 + ty + j, c = c0 + tx;
    if (c < C_DIM) tile[ty + j][tx] = x[((size_t)b * L_DIM + l) * C_DIM + c];
  }
  __syncthreads();
#pragma unroll
  for (int j = 0; j < 32; j += 8) {
    const int c = c0 + ty + j, l = l0 + tx;
    if (c < C_DIM)
      A0[((size_t)b * C_DIM + c) * L_DIM + l] = (bf16_t)tile[tx][ty + j];
  }
}

// LayerNorm over L=512 per row; optionally fuse multiply by xc (A0, bf16).
__global__ __launch_bounds__(256)
void layernorm_k(const float* __restrict__ in, const bf16_t* __restrict__ A0,
                 bf16_t* __restrict__ out, const float* __restrict__ gamma,
                 const float* __restrict__ beta, int mulA) {
  const int m = blockIdx.x;
  const int t = threadIdx.x;
  const float* row = in + (size_t)m * L_DIM;
  const float x0 = row[t];
  const float x1 = row[t + 256];
  __shared__ float rs[256], rs2[256];
  rs[t]  = x0 + x1;
  rs2[t] = x0 * x0 + x1 * x1;
  __syncthreads();
  for (int k = 128; k > 0; k >>= 1) {
    if (t < k) { rs[t] += rs[t + k]; rs2[t] += rs2[t + k]; }
    __syncthreads();
  }
  const float mean = rs[0] * (1.0f / L_DIM);
  const float var  = rs2[0] * (1.0f / L_DIM) - mean * mean;
  const float inv  = rsqrtf(var + 1e-6f);
  float y0 = (x0 - mean) * inv * gamma[t] + beta[t];
  float y1 = (x1 - mean) * inv * gamma[t + 256] + beta[t + 256];
  if (mulA) {
    y0 *= (float)A0[(size_t)m * L_DIM + t];
    y1 *= (float)A0[(size_t)m * L_DIM + t + 256];
  }
  out[(size_t)m * L_DIM + t]       = (bf16_t)y0;
  out[(size_t)m * L_DIM + t + 256] = (bf16_t)y1;
}

// ---------------------------------------------------------------------------
extern "C" void kernel_launch(void* const* d_in, const int* in_sizes, int n_in,
                              void* d_out, int out_size, void* d_ws, size_t ws_size,
                              hipStream_t stream) {
  const float* x     = (const float*)d_in[0];
  const float* W1    = (const float*)d_in[1];
  const float* W2    = (const float*)d_in[2];
  const float* gamma = (const float*)d_in[3];
  const float* beta  = (const float*)d_in[4];
  const float* Wl    = (const float*)d_in[5];
  const float* bl    = (const float*)d_in[6];
  float* out = (float*)d_out;

  const int M = M_ROWS;
  size_t off = 0;
  auto alloc = [&](size_t bytes) -> void* {
    void* p = (char*)d_ws + off;
    off += (bytes + 255) & ~(size_t)255;
    return p;
  };
  bf16_t* Dw   = (bf16_t*)alloc((size_t)L_DIM * L_DIM * 2);        // DCT, NxK
  bf16_t* W1b  = (bf16_t*)alloc((size_t)2 * L_DIM * L_DIM * 2);    // 1024x512
  bf16_t* W2b  = (bf16_t*)alloc((size_t)L_DIM * 2 * L_DIM * 2);    // 512x1024
  bf16_t* Wlb  = (bf16_t*)alloc((size_t)P_PAD * L_DIM * 2);        // 768x512
  bf16_t* A0   = (bf16_t*)alloc((size_t)M * L_DIM * 2);            // xc bf16
  bf16_t* Nbuf = (bf16_t*)alloc((size_t)M * L_DIM * 2);            // freqn / g
  bf16_t* Hbuf = (bf16_t*)alloc((size_t)M * 2 * L_DIM * 2);        // h
  float*  Fbuf = (float*) alloc((size_t)M * L_DIM * 4);            // freq / fw

  // Prep: DCT matrix + bf16 weights (+Wl zero-padded to 768 rows) + xc bf16.
  gen_dct_k<<<(L_DIM * L_DIM + 255) / 256, 256, 0, stream>>>(Dw);
  cvt_bf16_k<<<(2 * L_DIM * L_DIM + 255) / 256, 256, 0, stream>>>(
      W1, W1b, 2 * L_DIM * L_DIM);
  cvt_bf16_k<<<(2 * L_DIM * L_DIM + 255) / 256, 256, 0, stream>>>(
      W2, W2b, 2 * L_DIM * L_DIM);
  cvt_pad_bf16_k<<<(P_PAD * L_DIM + 255) / 256, 256, 0, stream>>>(
      Wl, Wlb, P_DIM, P_PAD, L_DIM);
  transpose_x_k<<<dim3(L_DIM / 32, (C_DIM + 31) / 32, B_DIM), dim3(32, 8),
                  0, stream>>>(x, A0);

  const int gmx = (M + 255) / 256;
  // Stage 1: freq = xc @ D^T  (f32 out)
  gemm_bf16_k<0><<<dim3(gmx, L_DIM / 64), 256, 0, stream>>>(
      A0, Dw, Fbuf, nullptr, M, L_DIM, L_DIM, nullptr, C_DIM, P_DIM);
  // LN(freq) -> bf16
  layernorm_k<<<M, 256, 0, stream>>>(Fbuf, A0, Nbuf, gamma, beta, 0);
  // Stage 2: h = relu(freqn @ W1^T) -> bf16
  gemm_bf16_k<1><<<dim3(gmx, 2 * L_DIM / 64), 256, 0, stream>>>(
      Nbuf, W1b, nullptr, Hbuf, M, 2 * L_DIM, L_DIM, nullptr, C_DIM, P_DIM);
  // Stage 3: fw = sigmoid(h @ W2^T) -> f32 (reuse Fbuf)
  gemm_bf16_k<2><<<dim3(gmx, L_DIM / 64), 256, 0, stream>>>(
      Hbuf, W2b, Fbuf, nullptr, M, L_DIM, 2 * L_DIM, nullptr, C_DIM, P_DIM);
  // g = xc * LN(fw) -> bf16 (reuse Nbuf)
  layernorm_k<<<M, 256, 0, stream>>>(Fbuf, A0, Nbuf, gamma, beta, 1);
  // Stage 4: out = g @ Wl^T + bl, transposed store to (B, P, C); B padded.
  gemm_bf16_k<3><<<dim3(gmx, P_PAD / 64), 256, 0, stream>>>(
      Nbuf, Wlb, out, nullptr, M, P_DIM, L_DIM, bl, C_DIM, P_DIM);
}
